// CausalSelfAttention_53197464928579
// MI455X (gfx1250) — matrix operand order repaired and checked by reading, not compiled, not run on hardware
//
#include <hip/hip_runtime.h>

// ---------------------------------------------------------------------------
// CDNA5 (gfx1250) causal self-attention.
// All matmuls on v_wmma_f32_16x16x32_bf16; GEMMs stage tiles in LDS via
// global_load_async_to_lds_b128 (ASYNCcnt) with double buffering.
// ---------------------------------------------------------------------------

typedef __bf16 bf_t;
typedef __attribute__((ext_vector_type(16))) __bf16 bf16x16;
typedef __attribute__((ext_vector_type(8)))  __bf16 bf16x8;
typedef __attribute__((ext_vector_type(8)))  float  f32x8;

#define B_  4
#define S_  2048
#define D_  2048
#define H_  16
#define KV_ 4
#define HD_ 128
#define EPS_ 1.1920929e-07f

#define BK_ 32          // k-chunk per stage
#define LDR_ 40         // LDS row stride in elements (80 B, multiple of 16 B)

__device__ __forceinline__ f32x8 wmma_bf16(bf16x16 a, bf16x16 b, f32x8 c) {
  // D = A(16x32) * B(32x16) + C, f32 accumulate
  return __builtin_amdgcn_wmma_f32_16x16x32_bf16(false, a, false, b, (short)0, c,
                                                 false, false);
}

__device__ __forceinline__ uint32_t lds_addr32(const void* p) {
  // Generic LDS pointer = {SHARED_BASE aperture, 32-bit LDS offset}; low 32
  // bits are the DS byte address.
  return (uint32_t)(uintptr_t)p;
}

__device__ __forceinline__ void async_copy_b128(uint32_t lds_dst, const void* gsrc) {
  // Per-lane 16B global -> LDS, tracked with ASYNCcnt.
  asm volatile("global_load_async_to_lds_b128 %0, %1, off"
               :: "v"(lds_dst), "v"((uint64_t)(uintptr_t)gsrc)
               : "memory");
}

__device__ __forceinline__ void wait_async0() {
  asm volatile("s_wait_asynccnt 0x0" ::: "memory");
}

// ---------------------------------------------------------------------------
// fp32 -> bf16 conversion
// ---------------------------------------------------------------------------
__global__ __launch_bounds__(256) void cvt_bf16_kernel(const float* __restrict__ src,
                                                       bf_t* __restrict__ dst,
                                                       size_t n) {
  size_t i = (size_t)blockIdx.x * blockDim.x + threadIdx.x;
  if (i < n) dst[i] = (bf_t)src[i];
}

// ---------------------------------------------------------------------------
// GEMM: Out[M,N] (f32) = A[M,K] (bf16 row-major) * W[N,K]^T (bf16 row-major)
// Block = 256 threads (8 waves, 4 M x 2 N), block tile 128x128, wave tile
// 32x64 (8 f32x8 accumulators). K staged in 32-chunks through double-buffered
// LDS via async copies; each wave reads fragments from LDS (A reused 4x,
// B reused 2x per stage).
// ---------------------------------------------------------------------------
__global__ __launch_bounds__(256) void gemm_bf16_wmma_lds(const bf_t* __restrict__ A,
                                                          const bf_t* __restrict__ W,
                                                          float* __restrict__ Out,
                                                          int M, int N, int K) {
  __shared__ bf_t Ash[2][128 * LDR_];
  __shared__ bf_t Bsh[2][128 * LDR_];

  const int tid   = threadIdx.x;
  const int lane  = tid & 31;
  const int wslot = tid >> 5;
  const int half  = lane >> 4;
  const int lr    = lane & 15;
  const int wy    = wslot >> 1;     // 0..3 : 32-row slice of M
  const int wx    = wslot & 1;      // 0..1 : 64-col slice of N

  const int nb   = N >> 7;
  const int mblk = blockIdx.x / nb;
  const int nblk = blockIdx.x % nb;
  const int m0   = mblk * 128;
  const int n0   = nblk * 128;
  (void)M;

  // Stage one 128x32 tile of A and of W into LDS buffer `buf` for k-base kb.
  // 128 rows x 64B = 512 x 16B chunks per matrix; 256 threads x 2 chunks.
  auto issue_stage = [&](int buf, int kb) {
    for (int j = 0; j < 2; ++j) {
      const int c   = j * 256 + tid;
      const int row = c >> 2;          // 0..127
      const int seg = (c & 3) * 16;    // byte segment within 64B row
      const uint32_t la = lds_addr32(&Ash[buf][row * LDR_]) + seg;
      const char* ga = (const char*)(A + (size_t)(m0 + row) * K + kb) + seg;
      async_copy_b128(la, ga);
      const uint32_t lb = lds_addr32(&Bsh[buf][row * LDR_]) + seg;
      const char* gb = (const char*)(W + (size_t)(n0 + row) * K + kb) + seg;
      async_copy_b128(lb, gb);
    }
  };

  f32x8 acc[2][4];
  for (int sm = 0; sm < 2; ++sm)
    for (int sn = 0; sn < 4; ++sn)
      for (int r = 0; r < 8; ++r) acc[sm][sn][r] = 0.0f;

  issue_stage(0, 0);
  wait_async0();
  __syncthreads();

  int buf = 0;
  for (int kb = 0; kb < K; kb += BK_) {
    if (kb + BK_ < K) issue_stage(buf ^ 1, kb + BK_);   // prefetch next chunk

    // A fragments (ISA 16-bit A layout): row per lane, k split by half.
    bf16x16 af[2];
    for (int sm = 0; sm < 2; ++sm) {
      const bf_t* ap = &Ash[buf][(wy * 32 + sm * 16 + lr) * LDR_ + half * 8];
      bf16x8 a0 = *(const bf16x8*)ap;
      bf16x8 a1 = *(const bf16x8*)(ap + 16);
      for (int j = 0; j < 8; ++j) { af[sm][j] = a0[j]; af[sm][8 + j] = a1[j]; }
    }

    // B fragments: output column per lane, k split by half (same order as A).
    for (int sn = 0; sn < 4; ++sn) {
      const bf_t* bp = &Bsh[buf][(wx * 64 + sn * 16 + lr) * LDR_ + half * 8];
      bf16x8 b0 = *(const bf16x8*)bp;
      bf16x8 b1 = *(const bf16x8*)(bp + 16);
      bf16x16 bfrag;
      for (int j = 0; j < 8; ++j) { bfrag[j] = b0[j]; bfrag[8 + j] = b1[j]; }
      acc[0][sn] = wmma_bf16(af[0], bfrag, acc[0][sn]);
      acc[1][sn] = wmma_bf16(af[1], bfrag, acc[1][sn]);
    }

    __syncthreads();            // all waves done reading `buf`
    if (kb + BK_ < K) {
      wait_async0();            // next buffer's async writes landed (this wave)
      __syncthreads();          // ... and for every wave in the block
    }
    buf ^= 1;
  }

  for (int sm = 0; sm < 2; ++sm)
    for (int sn = 0; sn < 4; ++sn)
      for (int r = 0; r < 8; ++r) {
        const int row = m0 + wy * 32 + sm * 16 + half * 8 + r;
        const int col = n0 + wx * 64 + sn * 16 + lr;
        Out[(size_t)row * N + col] = acc[sm][sn][r];
      }
}

// ---------------------------------------------------------------------------
// Fused per-head RMS norm + RoPE (first 64 dims) + optional gain.
// One wave per (b, s, head). Input: [B, S, nheads*128] f32 (GEMM output).
// Output: [B, nheads, S, 128] bf16.
// ---------------------------------------------------------------------------
__global__ __launch_bounds__(128) void norm_rope_kernel(const float* __restrict__ In,
                                                        bf_t* __restrict__ Outp,
                                                        const float* __restrict__ gain,
                                                        int nheads) {
  const int wave = blockIdx.x * (blockDim.x >> 5) + (threadIdx.x >> 5);
  const int lane = threadIdx.x & 31;

  const int h  = wave % nheads;
  const int bs = wave / nheads;            // = b*S + s
  const int s  = bs % S_;
  const int b  = bs / S_;
  if (b >= B_) return;

  const float* xp = In + (size_t)bs * (nheads * HD_) + h * HD_;
  float x0 = xp[lane];
  float x1 = xp[lane + 32];
  float x2 = xp[lane + 64];
  float x3 = xp[lane + 96];

  float ss = x0 * x0 + x1 * x1 + x2 * x2 + x3 * x3;
  for (int off = 1; off < 32; off <<= 1) ss += __shfl_xor(ss, off, 32);
  const float rn = rsqrtf(ss * (1.0f / (float)HD_) + EPS_);
  const float g = gain ? gain[h] : 1.0f;

  // RoPE: pair (d, d+32), d = lane in [0,32). inv_freq = 10000^(-2i/64)
  const float inv_freq = __expf(-((float)(2 * lane) / 64.0f) * 9.2103403719761836f);
  const float ang = (float)s * inv_freq;
  const float cs = __cosf(ang);
  const float sn = __sinf(ang);

  const float v0 = x0 * rn, v1 = x1 * rn;
  const float y0 = v0 * cs + v1 * sn;
  const float y1 = -v0 * sn + v1 * cs;

  bf_t* op = Outp + (((size_t)b * nheads + h) * S_ + s) * HD_;
  op[lane]      = (bf_t)(y0 * g);
  op[lane + 32] = (bf_t)(y1 * g);
  op[lane + 64] = (bf_t)(x2 * rn * g);
  op[lane + 96] = (bf_t)(x3 * rn * g);
}

// ---------------------------------------------------------------------------
// V transpose: Vf [B, S, KV*128] f32 -> Vt [B, KV, 128, S] bf16
// ---------------------------------------------------------------------------
__global__ __launch_bounds__(256) void vtrans_kernel(const float* __restrict__ Vf,
                                                     bf_t* __restrict__ Vt) {
  size_t i = (size_t)blockIdx.x * blockDim.x + threadIdx.x;
  const size_t n = (size_t)B_ * S_ * KV_ * HD_;
  if (i >= n) return;
  const int dd = (int)(i % (KV_ * HD_));
  const int bs = (int)(i / (KV_ * HD_));
  const int s  = bs % S_;
  const int b  = bs / S_;
  const int kv = dd / HD_;
  const int d  = dd % HD_;
  Vt[(((size_t)b * KV_ + kv) * HD_ + d) * S_ + s] = (bf_t)Vf[i];
}

// ---------------------------------------------------------------------------
// Flash attention (causal). One wave per 16 query rows of one (b, h).
// Qb [B,H,S,128] bf16, Kb [B,KV,S,128] bf16, Vt [B,KV,128,S] bf16,
// Yb [B,S,2048] bf16 (attention output, pre-projection).
// Per 32-key chunk: 8 WMMAs QK^T (K=128), online softmax, P via per-wave LDS
// (C-layout -> A-layout), 8 WMMAs P*V into 16x128 accumulator.
// No __syncthreads here: waves have different causal trip counts; per-wave
// LDS RAW is handled by in-order DS + compiler-inserted s_wait_dscnt.
// ---------------------------------------------------------------------------
__global__ __launch_bounds__(128) void attn_kernel(const bf_t* __restrict__ Qb,
                                                   const bf_t* __restrict__ Kb,
                                                   const bf_t* __restrict__ Vt,
                                                   bf_t* __restrict__ Yb) {
  __shared__ bf_t plds[4][16][48];   // per-wave 16x32 P tile, padded rows (96B)

  const int wslot = threadIdx.x >> 5;
  const int lane  = threadIdx.x & 31;
  const int half  = lane >> 4;
  const int lr    = lane & 15;

  const int wave = blockIdx.x * 4 + wslot;
  const int qt = wave % (S_ / 16);
  const int bh = wave / (S_ / 16);
  const int h  = bh % H_;
  const int b  = bh / H_;
  if (b >= B_) return;
  const int kvh = h / (H_ / KV_);

  // Q fragments: 4 chunks of 32 along head-dim, 16 contiguous k per half
  const bf_t* qrowp = Qb + (((size_t)b * H_ + h) * S_ + qt * 16 + lr) * HD_ + half * 16;
  bf16x16 qf[4];
  for (int cck = 0; cck < 4; ++cck) {
    bf16x8 a0 = *(const bf16x8*)(qrowp + cck * 32);
    bf16x8 a1 = *(const bf16x8*)(qrowp + cck * 32 + 8);
    for (int j = 0; j < 8; ++j) { qf[cck][j] = a0[j]; qf[cck][8 + j] = a1[j]; }
  }

  f32x8 acc[8];
  float mrow[8], lrow[8];
  for (int t = 0; t < 8; ++t)
    for (int r = 0; r < 8; ++r) acc[t][r] = 0.0f;
  for (int r = 0; r < 8; ++r) { mrow[r] = -3.0e38f; lrow[r] = 0.0f; }

  const bf_t* krow0 = Kb + ((size_t)b * KV_ + kvh) * S_ * HD_;
  const bf_t* vbase = Vt + ((size_t)b * KV_ + kvh) * HD_ * S_;   // [d][s]
  const float scale = 0.08838834764831845f;  // 1/sqrt(128)
  const int qend = qt * 16 + 15;

  for (int kb0 = 0; kb0 <= qend; kb0 += 32) {
    // ---- scores: two 16x16 tiles (key cols kb0..kb0+15 and kb0+16..kb0+31)
    f32x8 s0, s1;
    for (int r = 0; r < 8; ++r) { s0[r] = 0.0f; s1[r] = 0.0f; }
    for (int cck = 0; cck < 4; ++cck) {
      const bf_t* bp0 = krow0 + (size_t)(kb0 + lr) * HD_ + cck * 32 + half * 16;
      bf16x16 k0 = *(const bf16x16*)bp0;
      s0 = wmma_bf16(qf[cck], k0, s0);
      const bf_t* bp1 = krow0 + (size_t)(kb0 + 16 + lr) * HD_ + cck * 32 + half * 16;
      bf16x16 k1 = *(const bf16x16*)bp1;
      s1 = wmma_bf16(qf[cck], k1, s1);
    }
    if (kb0 + 32 <= qend)  // hint next K chunk into cache
      __builtin_prefetch(krow0 + (size_t)(kb0 + 32 + lr) * HD_, 0, 1);

    // ---- online softmax (row stats within each 16-lane half)
    for (int r = 0; r < 8; ++r) {
      const int qrow = qt * 16 + half * 8 + r;
      float v0 = s0[r] * scale;
      float v1 = s1[r] * scale;
      if (kb0 + lr > qrow)      v0 = -3.0e38f;
      if (kb0 + 16 + lr > qrow) v1 = -3.0e38f;

      float mx = fmaxf(v0, v1);
      for (int off = 1; off < 16; off <<= 1) mx = fmaxf(mx, __shfl_xor(mx, off, 32));
      const float mnew = fmaxf(mrow[r], mx);
      const float corr = __expf(mrow[r] - mnew);
      const float p0 = __expf(v0 - mnew);
      const float p1 = __expf(v1 - mnew);
      float ps = p0 + p1;
      for (int off = 1; off < 16; off <<= 1) ps += __shfl_xor(ps, off, 32);
      lrow[r] = lrow[r] * corr + ps;
      mrow[r] = mnew;
      for (int t = 0; t < 8; ++t) acc[t][r] *= corr;

      const int m = half * 8 + r;          // C-layout row -> LDS row
      plds[wslot][m][lr]      = (bf_t)p0;
      plds[wslot][m][16 + lr] = (bf_t)p1;
    }

    // ---- P fragment: read-back from this wave's LDS tile, 16 contiguous k per half.
    bf16x16 pf;
    for (int j = 0; j < 8; ++j) {
      pf[j]     = plds[wslot][lr][half * 16 + j];
      pf[8 + j] = plds[wslot][lr][half * 16 + 8 + j];
    }

    // ---- O += P(16x32) * V(32x16 per d-tile); V^T rows are contiguous in s
    for (int t = 0; t < 8; ++t) {
      const bf_t* vp = vbase + (size_t)(t * 16 + lr) * S_ + kb0 + half * 16;
      bf16x16 vfrag = *(const bf16x16*)vp;
      acc[t] = wmma_bf16(pf, vfrag, acc[t]);
    }
  }

  // ---- epilogue: normalize and store [b, qrow, h*128 + d] as bf16
  for (int t = 0; t < 8; ++t) {
    for (int r = 0; r < 8; ++r) {
      const int qrow = qt * 16 + half * 8 + r;
      const float o = acc[t][r] / lrow[r];
      Yb[((size_t)b * S_ + qrow) * D_ + h * HD_ + t * 16 + lr] = (bf_t)o;
    }
  }
}

// ---------------------------------------------------------------------------
// Host: orchestrate the pipeline on `stream` (graph-capture safe).
// ---------------------------------------------------------------------------
extern "C" void kernel_launch(void* const* d_in, const int* in_sizes, int n_in,
                              void* d_out, int out_size, void* d_ws, size_t ws_size,
                              hipStream_t stream) {
  (void)in_sizes; (void)n_in; (void)out_size; (void)ws_size;

  const float* x    = (const float*)d_in[0];
  const float* Wq   = (const float*)d_in[1];
  const float* Wk   = (const float*)d_in[2];
  const float* Wv   = (const float*)d_in[3];
  const float* Wp   = (const float*)d_in[4];
  const float* gain = (const float*)d_in[5];
  float* out = (float*)d_out;

  char* ws = (char*)d_ws;
  size_t off = 0;
  auto wsAlloc = [&](size_t bytes) -> void* {
    void* p = ws + off;
    off += (bytes + 255) & ~(size_t)255;
    return p;
  };

  const size_t NX  = (size_t)B_ * S_ * D_;          // 16,777,216
  const size_t NKV = (size_t)B_ * S_ * KV_ * HD_;   //  4,194,304
  const size_t NWQ = (size_t)D_ * D_;
  const size_t NWK = (size_t)KV_ * HD_ * D_;

  bf_t*  xb  = (bf_t*)wsAlloc(NX * 2);
  bf_t*  Wqb = (bf_t*)wsAlloc(NWQ * 2);
  bf_t*  Wkb = (bf_t*)wsAlloc(NWK * 2);
  bf_t*  Wvb = (bf_t*)wsAlloc(NWK * 2);
  bf_t*  Wpb = (bf_t*)wsAlloc(NWQ * 2);
  float* Qf  = (float*)wsAlloc(NX * 4);
  float* Kf  = (float*)wsAlloc(NKV * 4);
  float* Vf  = (float*)wsAlloc(NKV * 4);
  bf_t*  Qb  = (bf_t*)wsAlloc(NX * 2);
  bf_t*  Kb  = (bf_t*)wsAlloc(NKV * 2);
  bf_t*  Vt  = (bf_t*)wsAlloc(NKV * 2);
  bf_t*  Yb  = (bf_t*)wsAlloc(NX * 2);

  // 1) fp32 -> bf16 conversions
  cvt_bf16_kernel<<<(unsigned)((NX  + 255) / 256), 256, 0, stream>>>(x,  xb,  NX);
  cvt_bf16_kernel<<<(unsigned)((NWQ + 255) / 256), 256, 0, stream>>>(Wq, Wqb, NWQ);
  cvt_bf16_kernel<<<(unsigned)((NWK + 255) / 256), 256, 0, stream>>>(Wk, Wkb, NWK);
  cvt_bf16_kernel<<<(unsigned)((NWK + 255) / 256), 256, 0, stream>>>(Wv, Wvb, NWK);
  cvt_bf16_kernel<<<(unsigned)((NWQ + 255) / 256), 256, 0, stream>>>(Wp, Wpb, NWQ);

  // 2) QKV projections (async-LDS WMMA GEMM). M = B*S = 8192 rows.
  const int M = B_ * S_;
  gemm_bf16_wmma_lds<<<(M / 128) * (D_ / 128), 256, 0, stream>>>(
      xb, Wqb, Qf, M, D_, D_);                            // 1024 blocks
  gemm_bf16_wmma_lds<<<(M / 128) * ((KV_ * HD_) / 128), 256, 0, stream>>>(
      xb, Wkb, Kf, M, KV_ * HD_, D_);                     // 256 blocks
  gemm_bf16_wmma_lds<<<(M / 128) * ((KV_ * HD_) / 128), 256, 0, stream>>>(
      xb, Wvb, Vf, M, KV_ * HD_, D_);

  // 3) per-head RMS norm + RoPE (+gain for Q); V transpose to [B,KV,hd,S]
  norm_rope_kernel<<<(B_ * S_ * H_) / 4, 128, 0, stream>>>(Qf, Qb, gain, H_);
  norm_rope_kernel<<<(B_ * S_ * KV_) / 4, 128, 0, stream>>>(Kf, Kb, nullptr, KV_);
  vtrans_kernel<<<(unsigned)((NKV + 255) / 256), 256, 0, stream>>>(Vf, Vt);

  // 4) causal flash attention (WMMA for QK^T and P*V)
  attn_kernel<<<(B_ * H_ * (S_ / 16)) / 4, 128, 0, stream>>>(Qb, Kb, Vt, Yb);

  // 5) output projection -> fp32 d_out
  gemm_bf16_wmma_lds<<<(M / 128) * (D_ / 128), 256, 0, stream>>>(
      Yb, Wpb, out, M, D_, D_);
}
